// Conv4D_55405078118921
// MI455X (gfx1250) — matrix operand order repaired and checked by reading, not compiled, not run on hardware
//
#include <hip/hip_runtime.h>

// Conv4D as implicit GEMM on CDNA5 WMMA, bf16x3 split for fp32-class accuracy.
// M = 8*16^4 = 524288 output pixels, N = 64 couts, K = 81 taps * 32 cins.
// Block: 256 threads = 8 waves; block tile 128(M) x 64(N);
// wave tile 16(M) x 64(N) -> one A fetch feeds 12 WMMAs.
//
// One-time prepacks into d_ws (guarded by ws_size, uniform branches):
//   - weights: split {hi,lo} bf16 pairs in exact B-operand LDS layout (648 KB)
//   - x: separate bf16 hi / lo planes (33.5 MB each) -> A operands become
//     direct b128 loads with ZERO conversion VALU in the hot loop.
// Per tap, the 8 KB weight slice is streamed to LDS with double-buffered
// global_load_async_to_lds_b128 (ASYNCcnt), overlapping DMA with WMMA.

typedef __attribute__((ext_vector_type(16))) __bf16   v16bf;
typedef __attribute__((ext_vector_type(2)))  __bf16   v2bf;
typedef __attribute__((ext_vector_type(8)))  float    v8f;
typedef __attribute__((ext_vector_type(2)))  float    f2;
typedef __attribute__((ext_vector_type(8)))  unsigned v8u;
typedef __attribute__((ext_vector_type(4)))  unsigned u4;
typedef __attribute__((ext_vector_type(4)))  float    f4;

#define W_WORDS   ((size_t)81 * 2048)            // packed weight words
#define W_BYTES   (W_WORDS * 4)                  // 663552 B
#define XPLANE_W  ((size_t)8388608)              // 16.7M elems / 2 per word
#define WS_FULL   (W_BYTES + 2 * XPLANE_W * 4)   // + hi & lo planes (67.8 MB)

// Split two f32 into packed-pair bf16 hi word (.x) and residual-lo word (.y).
static __device__ __forceinline__ uint2 split_pair(float f0, float f1) {
  f2 f = {f0, f1};
  v2bf h  = __builtin_convertvector(f, v2bf);
  f2   hf = __builtin_convertvector(h, f2);
  f2   rs = f - hf;
  v2bf l  = __builtin_convertvector(rs, v2bf);
  return make_uint2(__builtin_bit_cast(unsigned, h),
                    __builtin_bit_cast(unsigned, l));
}

// ---- One-time weight prepack: W[t][cin][cout] f32 -> packed split words ----
// Wpk layout per tap t (2048 words): [0..1023] hi word (vp,cin) = vp*32+cin,
// [1024..2047] lo word. word(vp,cin) = {bf16 W[cin][2vp], bf16 W[cin][2vp+1]}.
__global__ __launch_bounds__(256)
void prepack_weights(const float* __restrict__ Wt, unsigned* __restrict__ Wpk) {
  const int p = blockIdx.x * 256 + threadIdx.x;   // 81*1024 = 82944 pairs
  if (p >= 81 * 1024) return;
  const int t  = p >> 10;
  const int r  = p & 1023;
  const int ci = r >> 5;
  const int vp = r & 31;
  const float* src = Wt + (size_t)t * 2048 + ci * 64 + 2 * vp;
  const uint2 s = split_pair(src[0], src[1]);
  Wpk[(size_t)t * 2048 + vp * 32 + ci]        = s.x;
  Wpk[(size_t)t * 2048 + 1024 + vp * 32 + ci] = s.y;
}

// ---- One-time x prepack: f32 [pix][32ch] -> bf16 hi plane + lo plane ----
// word w of a plane = channels {2w, 2w+1} of pixel w/16 (same flat order).
__global__ __launch_bounds__(256)
void prepack_x(const float* __restrict__ X, unsigned* __restrict__ Xhi,
               unsigned* __restrict__ Xlo) {
  const size_t p = (size_t)blockIdx.x * 256 + threadIdx.x;  // 4 elems each
  f4 v = *(const f4*)(X + p * 4);
  const uint2 s01 = split_pair(v[0], v[1]);
  const uint2 s23 = split_pair(v[2], v[3]);
  ((uint2*)Xhi)[p] = make_uint2(s01.x, s23.x);
  ((uint2*)Xlo)[p] = make_uint2(s01.y, s23.y);
}

__global__ __launch_bounds__(256)
void conv4d_wmma_bf16x3(const float* __restrict__ X,     // [8,16,16,16,16,32]
                        const float* __restrict__ Wt,    // [3,3,3,3,32,64]
                        const float* __restrict__ Bias,  // [64]
                        float* __restrict__ Out,         // [8,16,16,16,16,64]
                        const unsigned* __restrict__ Wpk,
                        const unsigned* __restrict__ Xhi,
                        const unsigned* __restrict__ Xlo,
                        int usePre, int useXpre) {
  // Double-buffered per-tap weight slice: [0..1023] hi, [1024..2047] lo.
  __shared__ unsigned ldsB[2 * 2048];    // 16 KB

  const int tid   = threadIdx.x;
  const int lane  = tid & 31;
  const int wave  = tid >> 5;
  const int mBase = blockIdx.x * 128 + wave * 16;

  // This lane's A-row (output pixel). A layout: lanes l and l+16 share M = l&15,
  // but hold different K halves (channel base +0 / +8 within each 16-K group).
  const int m  = mBase + (lane & 15);
  const int z0 = m & 15;
  const int y0 = (m >> 4) & 15;
  const int x0 = (m >> 8) & 15;
  const int w0 = (m >> 12) & 15;
  const int bb = m >> 16;
  const int cA = (lane & 16) ? 8 : 0;

  // Fallback weight-staging assignment: conflict-free LDS scatter (cin = lane).
  const int sCi = tid & 31;            // cin
  const int sCo = (tid >> 5) * 8;      // cout block of 8

  // LDS byte address of this thread's async-copy destination (8 words).
  const unsigned ldsDst0 = (unsigned)(uintptr_t)(&ldsB[tid * 8]);

  auto issue_copy = [&](int tap, int buf) {
    const unsigned* wp = Wpk + (size_t)tap * 2048 + tid * 8;
    asm volatile(
        "global_load_async_to_lds_b128 %0, %1, off\n\t"
        "global_load_async_to_lds_b128 %0, %1, off offset:16"
        :
        : "v"(ldsDst0 + (unsigned)buf * 8192),
          "v"((unsigned long long)(uintptr_t)wp)
        : "memory");
  };

  v8f acc0 = {}, acc1 = {}, acc2 = {}, acc3 = {};

  if (usePre) issue_copy(0, 0);        // prologue DMA for tap 0

  for (int t = 0; t < 81; ++t) {
    const int dz = (t % 3) - 1;
    const int dy = ((t / 3) % 3) - 1;
    const int dx = ((t / 9) % 3) - 1;
    const int dw = (t / 27) - 1;
    const int cur = t & 1;

    if (usePre) {
      asm volatile("s_wait_asynccnt 0" ::: "memory");  // own DMA complete
      __syncthreads();                 // all DMAs visible; prev reads done
      if (t + 1 < 81) issue_copy(t + 1, cur ^ 1);      // overlap with compute
    } else {
      __syncthreads();                 // previous iteration's LDS reads done
      // ---- Fallback: stage W[t][cin][cout], split + packed per B unit ----
      const float* wt = Wt + (size_t)t * 2048 + sCi * 64 + sCo;
      f4 wa = *(const f4*)(wt);
      f4 wb = *(const f4*)(wt + 4);
      const uint2 s0 = split_pair(wa[0], wa[1]);
      const uint2 s1 = split_pair(wa[2], wa[3]);
      const uint2 s2 = split_pair(wb[0], wb[1]);
      const uint2 s3 = split_pair(wb[2], wb[3]);
      const int vpBase = sCo >> 1;
      ldsB[(vpBase + 0) * 32 + sCi]        = s0.x;
      ldsB[1024 + (vpBase + 0) * 32 + sCi] = s0.y;
      ldsB[(vpBase + 1) * 32 + sCi]        = s1.x;
      ldsB[1024 + (vpBase + 1) * 32 + sCi] = s1.y;
      ldsB[(vpBase + 2) * 32 + sCi]        = s2.x;
      ldsB[1024 + (vpBase + 2) * 32 + sCi] = s2.y;
      ldsB[(vpBase + 3) * 32 + sCi]        = s3.x;
      ldsB[1024 + (vpBase + 3) * 32 + sCi] = s3.y;
      __syncthreads();
    }

    // ---- A tile: 16x32 gather with SAME zero padding ----
    const int iz = z0 + dz, iy = y0 + dy, ix = x0 + dx, iw = w0 + dw;
    const bool inb = ((unsigned)iz < 16u) && ((unsigned)iy < 16u) &&
                     ((unsigned)ix < 16u) && ((unsigned)iw < 16u);
    const size_t pix = ((((size_t)bb * 16 + iw) * 16 + ix) * 16 + iy) * 16 + iz;

    v16bf ah, al;
    if (useXpre) {
      // Pre-split planes: A operands are direct b128 loads, zero VALU.
      u4 h0 = {}, h1 = {}, l0 = {}, l1 = {};
      if (inb) {
        const unsigned* ph = Xhi + pix * 16 + (cA >> 1);
        const unsigned* pl = Xlo + pix * 16 + (cA >> 1);
        h0 = *(const u4*)(ph);        // channels cA..cA+7   (elems 0..7)
        h1 = *(const u4*)(ph + 8);    // channels cA+16..+23 (elems 8..15)
        l0 = *(const u4*)(pl);
        l1 = *(const u4*)(pl + 8);
      }
      v8u ahw, alw;
#pragma unroll
      for (int j = 0; j < 4; ++j) {
        ahw[j] = h0[j]; ahw[4 + j] = h1[j];
        alw[j] = l0[j]; alw[4 + j] = l1[j];
      }
      ah = __builtin_bit_cast(v16bf, ahw);
      al = __builtin_bit_cast(v16bf, alw);
    } else {
      f4 a0 = {}, a1 = {}, a2 = {}, a3 = {};
      if (inb) {
        const float* px = X + pix * 32;
        a0 = *(const f4*)(px + cA);
        a1 = *(const f4*)(px + cA + 4);
        a2 = *(const f4*)(px + cA + 16);
        a3 = *(const f4*)(px + cA + 20);
      }
      v8u ahw, alw;
      uint2 r;
      r = split_pair(a0[0], a0[1]); ahw[0] = r.x; alw[0] = r.y;
      r = split_pair(a0[2], a0[3]); ahw[1] = r.x; alw[1] = r.y;
      r = split_pair(a1[0], a1[1]); ahw[2] = r.x; alw[2] = r.y;
      r = split_pair(a1[2], a1[3]); ahw[3] = r.x; alw[3] = r.y;
      r = split_pair(a2[0], a2[1]); ahw[4] = r.x; alw[4] = r.y;
      r = split_pair(a2[2], a2[3]); ahw[5] = r.x; alw[5] = r.y;
      r = split_pair(a3[0], a3[1]); ahw[6] = r.x; alw[6] = r.y;
      r = split_pair(a3[2], a3[3]); ahw[7] = r.x; alw[7] = r.y;
      ah = __builtin_bit_cast(v16bf, ahw);
      al = __builtin_bit_cast(v16bf, alw);
    }
    // reconverged here: EXEC all-1s before WMMA

    // ---- All 4 N-subtiles against the same A: 12 WMMAs per tap ----
    const unsigned* bufB = ldsB + (usePre ? (cur * 2048) : 0);
#pragma unroll
    for (int nt = 0; nt < 4; ++nt) {
      v16bf bh, bl;
      const unsigned short* bufS = (const unsigned short*)bufB;
      const int nB = nt * 16 + (lane & 15);
#pragma unroll
      for (int s = 0; s < 16; ++s) {   // B elem s: N = nB, K = table order (matches A)
        const int k = ((s & 8) << 1) + ((lane & 16) ? 8 : 0) + (s & 7);
        const int idx = ((nB >> 1) * 32 + k) * 2 + (nB & 1);
        bh[s] = __builtin_bit_cast(__bf16, bufS[idx]);
        bl[s] = __builtin_bit_cast(__bf16, bufS[2048 + idx]);
      }
      v8f c = (nt == 0) ? acc0 : (nt == 1) ? acc1 : (nt == 2) ? acc2 : acc3;
      // bf16x3: Ah*Bh + Ah*Bl + Al*Bh  (drop lo*lo)
      c = __builtin_amdgcn_wmma_f32_16x16x32_bf16(false, ah, false, bh,
                                                  (short)0, c, false, false);
      c = __builtin_amdgcn_wmma_f32_16x16x32_bf16(false, ah, false, bl,
                                                  (short)0, c, false, false);
      c = __builtin_amdgcn_wmma_f32_16x16x32_bf16(false, al, false, bh,
                                                  (short)0, c, false, false);
      if      (nt == 0) acc0 = c;
      else if (nt == 1) acc1 = c;
      else if (nt == 2) acc2 = c;
      else              acc3 = c;
    }
  }

  // ---- Epilogue: (acc + bias) * 1/sqrt(2); 64B-contiguous per half-wave ----
  const int rBase = mBase + ((lane & 16) ? 8 : 0);
#pragma unroll
  for (int nt = 0; nt < 4; ++nt) {
    const int   n  = nt * 16 + (lane & 15);
    const float bv = Bias[n];
    const v8f   c  = (nt == 0) ? acc0 : (nt == 1) ? acc1 : (nt == 2) ? acc2 : acc3;
#pragma unroll
    for (int r = 0; r < 8; ++r) {
      Out[(size_t)(rBase + r) * 64 + n] = (c[r] + bv) * 0.70710678118654752440f;
    }
  }
}

extern "C" void kernel_launch(void* const* d_in, const int* in_sizes, int n_in,
                              void* d_out, int out_size, void* d_ws, size_t ws_size,
                              hipStream_t stream) {
  (void)in_sizes; (void)n_in; (void)out_size;
  const float* X    = (const float*)d_in[0];
  const float* Wt   = (const float*)d_in[1];
  const float* Bias = (const float*)d_in[2];
  float*       Out  = (float*)d_out;

  const int usePre  = (d_ws != nullptr && ws_size >= W_BYTES) ? 1 : 0;
  const int useXpre = (d_ws != nullptr && ws_size >= WS_FULL) ? 1 : 0;
  unsigned* Wpk = (unsigned*)d_ws;
  unsigned* Xhi = (unsigned*)((char*)d_ws + W_BYTES);
  unsigned* Xlo = Xhi + XPLANE_W;

  if (usePre) {
    // 82944 cout-pairs: one-time split/pack of the 648 KB weight tensor.
    prepack_weights<<<dim3(324), dim3(256), 0, stream>>>(Wt, Wpk);
  }
  if (useXpre) {
    // 16.7M elements, 4 per thread: split x into bf16 hi/lo planes.
    prepack_x<<<dim3(16384), dim3(256), 0, stream>>>(X, Xhi, Xlo);
  }

  // M = 524288 pixels / 128 per block = 4096 blocks of 256 threads (8 waves)
  conv4d_wmma_bf16x3<<<dim3(4096), dim3(256), 0, stream>>>(
      X, Wt, Bias, Out, Wpk, Xhi, Xlo, usePre, useXpre);
}
